// TransformerEncoder_32676111188662
// MI455X (gfx1250) — matrix operand order repaired and checked by reading, not compiled, not run on hardware
//
#include <hip/hip_runtime.h>
#include <math.h>

// ---------------------------------------------------------------------------
// Types
// ---------------------------------------------------------------------------
typedef __bf16         v16bf  __attribute__((ext_vector_type(16)));
typedef float          v8f    __attribute__((ext_vector_type(8)));
typedef unsigned int   u32x4  __attribute__((ext_vector_type(4)));
typedef float          f32x4  __attribute__((ext_vector_type(4)));
typedef unsigned short u16x4  __attribute__((ext_vector_type(4)));

union FragBf { v16bf v; u32x4 q[2]; };

#define F_DIM   1024
#define HEADS   16
#define HDIM    64
#define BATCH   4
#define SEQ     1024
#define ROWS    (BATCH * SEQ)      // 4096 tokens
#define MLP_H   4096
#define ATT_SCALE 0.03125f         // 1/sqrt(1024)

__device__ inline unsigned short f2bf(float f) {
    union { float f; unsigned u; } v; v.f = f;
    unsigned r = v.u + 0x7FFFu + ((v.u >> 16) & 1u);   // round-to-nearest-even
    return (unsigned short)(r >> 16);
}

__device__ inline v8f wmma_bf16(v16bf a, v16bf b, v8f c) {
    return __builtin_amdgcn_wmma_f32_16x16x32_bf16(false, a, false, b, (short)0, c,
                                                   false, false);
}

__device__ inline v8f zero8() {
    v8f z;
#pragma unroll
    for (int i = 0; i < 8; ++i) z[i] = 0.0f;
    return z;
}

__device__ inline float gelu_exact(float x) {
    return 0.5f * x * (1.0f + erff(x * 0.70710678118654752f));
}

// ---------------------------------------------------------------------------
// CDNA5 async memory->LDS copy (ASYNCcnt-tracked, no VGPR round trip).
// ---------------------------------------------------------------------------
__device__ inline void async_ld_b128(unsigned lds_off, const void* gaddr) {
    asm volatile("global_load_async_to_lds_b128 %0, %1, off"
                 :: "v"(lds_off), "v"((unsigned long long)(uintptr_t)gaddr)
                 : "memory");
}
__device__ inline void wait_async0() {
    asm volatile("s_wait_asynccnt 0x0" ::: "memory");
}
__device__ inline void wait_async4() {
    asm volatile("s_wait_asynccnt 0x4" ::: "memory");
}
__device__ inline void wait_async8() {
    asm volatile("s_wait_asynccnt 0x8" ::: "memory");
}

// ---------------------------------------------------------------------------
// Fused fp32 -> bf16 transpose: in[K][N] f32  ->  out[N][K] bf16.
// ---------------------------------------------------------------------------
__global__ __launch_bounds__(256) void cvt_tr_kernel(const float* __restrict__ in,
                                                     unsigned short* __restrict__ out,
                                                     int K, int N) {
    __shared__ unsigned short t[32][33];
    const int n0 = blockIdx.x * 32;
    const int k0 = blockIdx.y * 32;
    const int tx = threadIdx.x & 31;
    const int ty = threadIdx.x >> 5;      // 0..7
#pragma unroll
    for (int i = 0; i < 32; i += 8)
        t[ty + i][tx] = f2bf(in[(size_t)(k0 + ty + i) * N + n0 + tx]);
    __syncthreads();
#pragma unroll
    for (int i = 0; i < 32; i += 8)
        out[(size_t)(n0 + ty + i) * K + k0 + tx] = t[tx][ty + i];
}

// ---------------------------------------------------------------------------
// LayerNorm (one block per row of 1024) -> bf16
// ---------------------------------------------------------------------------
__global__ __launch_bounds__(256) void ln_bf16_kernel(const float* __restrict__ x,
                                                      const float* __restrict__ g,
                                                      const float* __restrict__ b,
                                                      unsigned short* __restrict__ out) {
    const int row = blockIdx.x;
    const int tid = threadIdx.x;
    const float* xr = x + (size_t)row * F_DIM;

    f32x4 v = *(const f32x4*)(xr + tid * 4);
    float s  = v[0] + v[1] + v[2] + v[3];
    float s2 = v[0]*v[0] + v[1]*v[1] + v[2]*v[2] + v[3]*v[3];
#pragma unroll
    for (int m = 1; m < 32; m <<= 1) {
        s  += __shfl_xor(s,  m, 32);
        s2 += __shfl_xor(s2, m, 32);
    }
    __shared__ float ws[8], ws2[8];
    __shared__ float mu_s, rs_s;
    int lane = tid & 31, wid = tid >> 5;
    if (lane == 0) { ws[wid] = s; ws2[wid] = s2; }
    __syncthreads();
    if (tid == 0) {
        float ts = 0.f, ts2 = 0.f;
#pragma unroll
        for (int i = 0; i < 8; ++i) { ts += ws[i]; ts2 += ws2[i]; }
        float mu  = ts * (1.0f / F_DIM);
        float var = ts2 * (1.0f / F_DIM) - mu * mu;
        mu_s = mu;
        rs_s = rsqrtf(var + 1e-5f);
    }
    __syncthreads();
    float mu = mu_s, rs = rs_s;
    u16x4 o;
#pragma unroll
    for (int i = 0; i < 4; ++i) {
        int c = tid * 4 + i;
        o[i] = f2bf((v[i] - mu) * rs * g[c] + b[c]);
    }
    *(u16x4*)(out + (size_t)row * F_DIM + tid * 4) = o;
}

// ---------------------------------------------------------------------------
// Tiled bf16 WMMA GEMM:  C[M,N] = A[M,K] * Bt[N,K]^T + bias,  fused epilogues.
//   mode 0: QK   -> bf16 out remapped to [b, h, n, d]
//   mode 1: Oproj-> f32 out = C + bias + res
//   mode 2: MLP1 -> bf16 out = gelu(C + bias)
//   mode 3: MLP2 -> f32 out = gelu(C + bias) + res
//   mode 4: Vt   -> bf16 out remapped to [b, h, d, n]; bias indexed by ROW
// 256 threads = 8 waves (4x2), block tile 128x128, wave tile 32x64,
// K-step 32, double-buffered straight-line async copies, peeled last step.
// ---------------------------------------------------------------------------
#define GS 40   // LDS row stride (elements) -> 80B rows, 16B-aligned frag chunks

__global__ __launch_bounds__(256) void gemm_bf16_kernel(
    const unsigned short* __restrict__ A,   // [M,K] bf16
    const unsigned short* __restrict__ Bt,  // [N,K] bf16 (pre-transposed)
    const float* __restrict__ bias,
    const float* __restrict__ res,
    float* __restrict__ outf,
    unsigned short* __restrict__ outb,
    int M, int N, int K, int mode) {

    __shared__ __align__(16) unsigned short As[2][128 * GS];
    __shared__ __align__(16) unsigned short Bs[2][128 * GS];

    const int tid   = threadIdx.x;
    const int lane  = tid & 31;
    const int laneN = lane & 15;
    const int hi    = lane >> 4;
    const int wid   = tid >> 5;
    const int wrow  = wid >> 1;           // 0..3 -> 32-row strip
    const int wcol  = wid & 1;            // 0..1 -> 64-col strip
    const int m0    = blockIdx.y * 128;
    const int n0    = blockIdx.x * 128;

    const unsigned aB0 = (unsigned)(uintptr_t)&As[0][0];
    const unsigned aB1 = (unsigned)(uintptr_t)&As[1][0];
    const unsigned bB0 = (unsigned)(uintptr_t)&Bs[0][0];
    const unsigned bB1 = (unsigned)(uintptr_t)&Bs[1][0];

    v8f acc[2][4];
#pragma unroll
    for (int s = 0; s < 2; ++s)
#pragma unroll
        for (int f = 0; f < 4; ++f) acc[s][f] = zero8();

    const int koffA = hi ? 8 : 0;
    const int koffB = hi ? 8 : 0;

    // straight-line copy coordinates: each thread owns rows cr and cr+64
    const int cr = tid >> 2;
    const int ck = (tid & 3) * 8;
    const unsigned ldsLo = (unsigned)((cr * GS + ck) * 2);
    const unsigned ldsHi = (unsigned)(((cr + 64) * GS + ck) * 2);

    auto issue_tile = [&](int k0, int buf) {
        const unsigned ab = buf ? aB1 : aB0;
        const unsigned bb = buf ? bB1 : bB0;
        const unsigned short* ap = A  + (size_t)(m0 + cr) * K + k0 + ck;
        const unsigned short* bp = Bt + (size_t)(n0 + cr) * K + k0 + ck;
        async_ld_b128(ab + ldsLo, ap);
        async_ld_b128(ab + ldsHi, ap + (size_t)64 * K);
        async_ld_b128(bb + ldsLo, bp);
        async_ld_b128(bb + ldsHi, bp + (size_t)64 * K);
    };

    auto compute = [&](int cur) {
        FragBf afr[2], bfr[4];
#pragma unroll
        for (int s = 0; s < 2; ++s) {
            int r = wrow * 32 + s * 16 + laneN;
            afr[s].q[0] = *(const u32x4*)(&As[cur][r * GS + koffA]);
            afr[s].q[1] = *(const u32x4*)(&As[cur][r * GS + koffA + 16]);
        }
#pragma unroll
        for (int f = 0; f < 4; ++f) {
            int col = wcol * 64 + f * 16 + laneN;
            bfr[f].q[0] = *(const u32x4*)(&Bs[cur][col * GS + koffB]);
            bfr[f].q[1] = *(const u32x4*)(&Bs[cur][col * GS + koffB + 16]);
        }
#pragma unroll
        for (int f = 0; f < 4; ++f)
#pragma unroll
            for (int s = 0; s < 2; ++s)
                acc[s][f] = wmma_bf16(afr[s].v, bfr[f].v, acc[s][f]);
    };

    const int nk = K >> 5;
    issue_tile(0, 0);
    for (int ki = 0; ki < nk - 1; ++ki) {          // steady state: always prefetch
        const int cur = ki & 1;
        issue_tile((ki + 1) << 5, cur ^ 1);
        wait_async4();
        __syncthreads();
        compute(cur);
        __syncthreads();
    }
    wait_async0();                                  // peeled last K-step
    __syncthreads();
    compute((nk - 1) & 1);

    // --- epilogue ---
#pragma unroll
    for (int s = 0; s < 2; ++s) {
#pragma unroll
        for (int f = 0; f < 4; ++f) {
            int col = n0 + wcol * 64 + f * 16 + laneN;
            float bcol = (mode == 4) ? 0.0f : bias[col];
#pragma unroll
            for (int r = 0; r < 8; ++r) {
                int row = m0 + wrow * 32 + s * 16 + r + hi * 8;
                float v = acc[s][f][r] + ((mode == 4) ? bias[row] : bcol);
                if (mode == 0) {            // [token, feat] -> [b,h,n,d]
                    int bb2 = row >> 10, n = row & 1023;
                    int h   = col >> 6,  d = col & 63;
                    size_t idx = ((((size_t)bb2 * HEADS + h) * SEQ) + n) * HDIM + d;
                    outb[idx] = f2bf(v);
                } else if (mode == 1) {
                    size_t idx = (size_t)row * N + col;
                    outf[idx] = v + res[idx];
                } else if (mode == 2) {
                    outb[(size_t)row * N + col] = f2bf(gelu_exact(v));
                } else if (mode == 3) {
                    size_t idx = (size_t)row * N + col;
                    outf[idx] = gelu_exact(v) + res[idx];
                } else {                    // mode 4: [feat, token] -> [b,h,d,n]
                    int h   = row >> 6,  d = row & 63;
                    int bb2 = col >> 10, n = col & 1023;
                    size_t idx = ((((size_t)bb2 * HEADS + h) * HDIM) + d) * SEQ + n;
                    outb[idx] = f2bf(v);
                }
            }
        }
    }
}

// ---------------------------------------------------------------------------
// Flash attention: block = (b*h, 64-query tile), 128 threads = 4 waves.
// Q/K bf16 [B*H, SEQ, 64]; V pre-transposed bf16 [B*H, 64, SEQ].
// Double-buffered async K/Vt tiles, peeled last tile, online softmax.
// ---------------------------------------------------------------------------
#define AS 72   // LDS row stride for 64-wide tiles

__global__ __launch_bounds__(128) void attn_kernel(
    const unsigned short* __restrict__ Q,
    const unsigned short* __restrict__ Km,
    const unsigned short* __restrict__ Vt,
    unsigned short* __restrict__ Aout) {

    __shared__ __align__(16) unsigned short Ks [2][64 * AS];
    __shared__ __align__(16) unsigned short Vts[2][64 * AS];
    __shared__ __align__(16) unsigned short Ps [4][16 * AS];

    const int bh    = blockIdx.x;          // 0..63
    const int qt    = blockIdx.y;          // 0..15
    const int bb    = bh >> 4, h = bh & 15;
    const int tid   = threadIdx.x;
    const int lane  = tid & 31;
    const int laneN = lane & 15;
    const int hi    = lane >> 4;
    const int wid   = tid >> 5;            // 0..3

    const size_t base  = (size_t)bh * SEQ * HDIM;   // Q / K base
    const size_t baseV = (size_t)bh * HDIM * SEQ;   // Vt base: [d][n]

    const unsigned kB0 = (unsigned)(uintptr_t)&Ks[0][0];
    const unsigned kB1 = (unsigned)(uintptr_t)&Ks[1][0];
    const unsigned vB0 = (unsigned)(uintptr_t)&Vts[0][0];
    const unsigned vB1 = (unsigned)(uintptr_t)&Vts[1][0];

    // Q fragments for this wave's 16 rows (held for the whole kernel)
    FragBf qfr[2];
    {
        int qrow = qt * 64 + wid * 16 + laneN;
        const unsigned short* qp = Q + base + (size_t)qrow * HDIM;
        int ka = hi * 8;
        qfr[0].q[0] = *(const u32x4*)(qp + ka);
        qfr[0].q[1] = *(const u32x4*)(qp + ka + 16);
        qfr[1].q[0] = *(const u32x4*)(qp + 32 + ka);
        qfr[1].q[1] = *(const u32x4*)(qp + 32 + ka + 16);
    }

    // straight-line copy coordinates: each thread owns rows cr+16*i
    const int cr = tid >> 3;
    const int ck = (tid & 7) * 8;

    auto issue_tile = [&](int kt, int buf) {
        const unsigned kb = buf ? kB1 : kB0;
        const unsigned vb = buf ? vB1 : vB0;
        const unsigned short* kp = Km + base + (size_t)(kt * 64 + cr) * HDIM + ck;
        const unsigned short* vp = Vt + baseV + (size_t)cr * SEQ + kt * 64 + ck;
#pragma unroll
        for (int i = 0; i < 4; ++i)
            async_ld_b128(kb + (unsigned)(((cr + 16 * i) * AS + ck) * 2),
                          kp + (size_t)(16 * i) * HDIM);
#pragma unroll
        for (int i = 0; i < 4; ++i)
            async_ld_b128(vb + (unsigned)(((cr + 16 * i) * AS + ck) * 2),
                          vp + (size_t)(16 * i) * SEQ);
    };

    float mrow[8], lrow[8];
    v8f acco[4];
#pragma unroll
    for (int r = 0; r < 8; ++r) { mrow[r] = -3.0e38f; lrow[r] = 0.0f; }
#pragma unroll
    for (int f = 0; f < 4; ++f) acco[f] = zero8();

    auto tile_body = [&](int cur) {
        // ---- S = Q * K^T ----
        v8f accs[4];
#pragma unroll
        for (int f = 0; f < 4; ++f) accs[f] = zero8();
#pragma unroll
        for (int f = 0; f < 4; ++f) {
            int krow = f * 16 + laneN;
            FragBf k0, k1;
            k0.q[0] = *(const u32x4*)(&Ks[cur][krow * AS + hi * 8]);
            k0.q[1] = *(const u32x4*)(&Ks[cur][krow * AS + hi * 8 + 16]);
            k1.q[0] = *(const u32x4*)(&Ks[cur][krow * AS + 32 + hi * 8]);
            k1.q[1] = *(const u32x4*)(&Ks[cur][krow * AS + 32 + hi * 8 + 16]);
            accs[f] = wmma_bf16(qfr[0].v, k0.v, accs[f]);
            accs[f] = wmma_bf16(qfr[1].v, k1.v, accs[f]);
        }

        // ---- online softmax ----
        float mloc[8], alpha[8], lloc[8];
#pragma unroll
        for (int r = 0; r < 8; ++r) mloc[r] = -3.0e38f;
#pragma unroll
        for (int f = 0; f < 4; ++f)
#pragma unroll
            for (int r = 0; r < 8; ++r)
                mloc[r] = fmaxf(mloc[r], accs[f][r] * ATT_SCALE);
#pragma unroll
        for (int r = 0; r < 8; ++r) {
#pragma unroll
            for (int m = 1; m < 16; m <<= 1)
                mloc[r] = fmaxf(mloc[r], __shfl_xor(mloc[r], m, 32));
            float mn = fmaxf(mrow[r], mloc[r]);
            alpha[r] = __expf(mrow[r] - mn);
            mrow[r]  = mn;
            lrow[r] *= alpha[r];
            lloc[r]  = 0.0f;
        }
#pragma unroll
        for (int f = 0; f < 4; ++f)
#pragma unroll
            for (int r = 0; r < 8; ++r) {
                float p = __expf(accs[f][r] * ATT_SCALE - mrow[r]);
                lloc[r] += p;
                Ps[wid][(r + hi * 8) * AS + f * 16 + laneN] = f2bf(p);
            }
#pragma unroll
        for (int r = 0; r < 8; ++r) {
#pragma unroll
            for (int m = 1; m < 16; m <<= 1)
                lloc[r] += __shfl_xor(lloc[r], m, 32);
            lrow[r] += lloc[r];
        }
#pragma unroll
        for (int f = 0; f < 4; ++f)
#pragma unroll
            for (int r = 0; r < 8; ++r) acco[f][r] *= alpha[r];

        asm volatile("s_wait_dscnt 0x0" ::: "memory");  // P stores -> gathers

        // ---- O += P * V ----
        FragBf pfr[2];
        {
            int ka = hi * 8;
            pfr[0].q[0] = *(const u32x4*)(&Ps[wid][laneN * AS + ka]);
            pfr[0].q[1] = *(const u32x4*)(&Ps[wid][laneN * AS + ka + 16]);
            pfr[1].q[0] = *(const u32x4*)(&Ps[wid][laneN * AS + 32 + ka]);
            pfr[1].q[1] = *(const u32x4*)(&Ps[wid][laneN * AS + 32 + ka + 16]);
        }
#pragma unroll
        for (int f = 0; f < 4; ++f) {
            int drow = f * 16 + laneN;
            FragBf v0, v1;
            v0.q[0] = *(const u32x4*)(&Vts[cur][drow * AS + hi * 8]);
            v0.q[1] = *(const u32x4*)(&Vts[cur][drow * AS + hi * 8 + 16]);
            v1.q[0] = *(const u32x4*)(&Vts[cur][drow * AS + 32 + hi * 8]);
            v1.q[1] = *(const u32x4*)(&Vts[cur][drow * AS + 32 + hi * 8 + 16]);
            acco[f] = wmma_bf16(pfr[0].v, v0.v, acco[f]);
            acco[f] = wmma_bf16(pfr[1].v, v1.v, acco[f]);
        }
    };

    issue_tile(0, 0);
    for (int kt = 0; kt < 15; ++kt) {               // steady state
        const int cur = kt & 1;
        issue_tile(kt + 1, cur ^ 1);
        wait_async8();
        __syncthreads();
        tile_body(cur);
        __syncthreads();
    }
    wait_async0();                                  // peeled last tile
    __syncthreads();
    tile_body(15 & 1);

    // ---- normalize + store: attn[b, i, h*64+d] ----
#pragma unroll
    for (int f = 0; f < 4; ++f)
#pragma unroll
        for (int r = 0; r < 8; ++r) {
            int i = qt * 64 + wid * 16 + r + hi * 8;
            int d = f * 16 + laneN;
            float o = acco[f][r] / lrow[r];
            Aout[((size_t)bb * SEQ + i) * F_DIM + h * HDIM + d] = f2bf(o);
        }
}

// ---------------------------------------------------------------------------
// Host-side orchestration
// ---------------------------------------------------------------------------
extern "C" void kernel_launch(void* const* d_in, const int* in_sizes, int n_in,
                              void* d_out, int out_size, void* d_ws, size_t ws_size,
                              hipStream_t stream) {
    const float* x     = (const float*)d_in[0];
    const float* wq    = (const float*)d_in[1];
    const float* bq    = (const float*)d_in[2];
    const float* wk    = (const float*)d_in[3];
    const float* bk    = (const float*)d_in[4];
    const float* wv    = (const float*)d_in[5];
    const float* bv    = (const float*)d_in[6];
    const float* wo    = (const float*)d_in[7];
    const float* bo    = (const float*)d_in[8];
    const float* ln1g  = (const float*)d_in[9];
    const float* ln1b  = (const float*)d_in[10];
    const float* ln2g  = (const float*)d_in[11];
    const float* ln2b  = (const float*)d_in[12];
    const float* w1    = (const float*)d_in[13];
    const float* b1    = (const float*)d_in[14];
    const float* w2    = (const float*)d_in[15];
    const float* b2    = (const float*)d_in[16];
    float* out = (float*)d_out;

    char* ws = (char*)d_ws;
    auto carve = [&](size_t bytes) {
        char* p = ws;
        ws += (bytes + 255) & ~(size_t)255;
        return p;
    };
    const size_t FF  = (size_t)F_DIM * F_DIM;
    const size_t FH  = (size_t)F_DIM * MLP_H;
    const size_t RF  = (size_t)ROWS * F_DIM;
    const size_t RH  = (size_t)ROWS * MLP_H;

    unsigned short* wqt  = (unsigned short*)carve(FF * 2);   // [N][K] bf16
    unsigned short* wkt  = (unsigned short*)carve(FF * 2);
    unsigned short* wvt  = (unsigned short*)carve(FF * 2);
    unsigned short* wot  = (unsigned short*)carve(FF * 2);
    unsigned short* w1t  = (unsigned short*)carve(FH * 2);   // [4096][1024]
    unsigned short* w2t  = (unsigned short*)carve(FH * 2);   // [1024][4096]
    unsigned short* xnb  = (unsigned short*)carve(RF * 2);
    unsigned short* qb   = (unsigned short*)carve(RF * 2);   // [b,h,n,d]
    unsigned short* kb   = (unsigned short*)carve(RF * 2);   // [b,h,n,d]
    unsigned short* vtb  = (unsigned short*)carve(RF * 2);   // [b,h,d,n]
    unsigned short* atb  = (unsigned short*)carve(RF * 2);
    float*          ores = (float*)carve(RF * 4);
    unsigned short* ln2o = (unsigned short*)carve(RF * 2);
    unsigned short* h1b  = (unsigned short*)carve(RH * 2);

    // 1) weights -> bf16, transposed to [N][K]
    dim3 gT1(F_DIM / 32, F_DIM / 32);
    cvt_tr_kernel<<<gT1, 256, 0, stream>>>(wq, wqt, F_DIM, F_DIM);
    cvt_tr_kernel<<<gT1, 256, 0, stream>>>(wk, wkt, F_DIM, F_DIM);
    cvt_tr_kernel<<<gT1, 256, 0, stream>>>(wv, wvt, F_DIM, F_DIM);
    cvt_tr_kernel<<<gT1, 256, 0, stream>>>(wo, wot, F_DIM, F_DIM);
    dim3 gTw1(MLP_H / 32, F_DIM / 32);
    cvt_tr_kernel<<<gTw1, 256, 0, stream>>>(w1, w1t, F_DIM, MLP_H);
    dim3 gTw2(F_DIM / 32, MLP_H / 32);
    cvt_tr_kernel<<<gTw2, 256, 0, stream>>>(w2, w2t, MLP_H, F_DIM);

    // 2) LN1: x -> xn (bf16)
    ln_bf16_kernel<<<ROWS, 256, 0, stream>>>(x, ln1g, ln1b, xnb);

    // 3) Q/K projections (mode 0), V^T projection (mode 4: A=Wv^T, Bt=Xn)
    dim3 gQK(F_DIM / 128, ROWS / 128);
    gemm_bf16_kernel<<<gQK, 256, 0, stream>>>(xnb, wqt, bq, nullptr, nullptr, qb,
                                              ROWS, F_DIM, F_DIM, 0);
    gemm_bf16_kernel<<<gQK, 256, 0, stream>>>(xnb, wkt, bk, nullptr, nullptr, kb,
                                              ROWS, F_DIM, F_DIM, 0);
    dim3 gV(ROWS / 128, F_DIM / 128);
    gemm_bf16_kernel<<<gV, 256, 0, stream>>>(wvt, xnb, bv, nullptr, nullptr, vtb,
                                             F_DIM, ROWS, F_DIM, 4);

    // 4) flash attention -> attn (bf16, [b,n,f])
    dim3 gAtt(BATCH * HEADS, SEQ / 64);
    attn_kernel<<<gAtt, 128, 0, stream>>>(qb, kb, vtb, atb);

    // 5) O projection + residual (mode 1) -> ores (f32)
    gemm_bf16_kernel<<<gQK, 256, 0, stream>>>(atb, wot, bo, x, ores, nullptr,
                                              ROWS, F_DIM, F_DIM, 1);

    // 6) LN2: ores -> ln2o (bf16)
    ln_bf16_kernel<<<ROWS, 256, 0, stream>>>(ores, ln2g, ln2b, ln2o);

    // 7) MLP up + gelu (mode 2) -> h1 (bf16)
    dim3 gM1(MLP_H / 128, ROWS / 128);
    gemm_bf16_kernel<<<gM1, 256, 0, stream>>>(ln2o, w1t, b1, nullptr, nullptr, h1b,
                                              ROWS, MLP_H, F_DIM, 2);

    // 8) MLP down + gelu + residual (mode 3) -> d_out (f32)
    dim3 gM2(F_DIM / 128, ROWS / 128);
    gemm_bf16_kernel<<<gM2, 256, 0, stream>>>(h1b, w2t, b2, ores, out, nullptr,
                                              ROWS, F_DIM, MLP_H, 3);
}
